// TransformerEncoderLayer_36103495090564
// MI455X (gfx1250) — matrix operand order repaired and checked by reading, not compiled, not run on hardware
//
#include <hip/hip_runtime.h>
#include <hip/hip_bf16.h>

// ---------------------------------------------------------------------------
// Types matching the gfx1250 WMMA builtin signatures
// ---------------------------------------------------------------------------
typedef __bf16 bf16_t;
typedef __attribute__((ext_vector_type(16))) __bf16 v16bf;
typedef __attribute__((ext_vector_type(8)))  __bf16 v8bf;
typedef __attribute__((ext_vector_type(8)))  float  v8f;

#define D_MODEL 768
#define SEQ     2048
#define BATCH   8
#define ROWS    (BATCH * SEQ)       // 16384
#define LN_EPS  1e-5f
#define ATT_SCALE 0.03608439182435161f  // 1/sqrt(768)

// ---------------------------------------------------------------------------
// WMMA fragment helpers (CDNA5 wave32 layouts, CDNA5 ISA §7.12.2)
//
// A-matrix 16x32 (MxK), 16-bit: lane L(0..15) holds M=L; element i lives in
// VGPR i/2; K(i,lane) = (i/8)*16 + (lane/16)*8 + (i%8)  -> two contiguous
// 8-element (16 B) chunks per lane at k offsets (lane>>4)*8 and 16+(lane>>4)*8.
//
// B-matrix 32x16 (KxN), 16-bit: lane holds column N=lane&15; same K order as A,
// K(i,lane) = (i/8)*16 + (lane/16)*8 + (i%8) -> two 8-element chunks per lane when
// the operand is stored "Bt[n][k]" (n-major, k contiguous).
// ---------------------------------------------------------------------------
__device__ __forceinline__ v16bf load_frag_a(const bf16_t* __restrict__ base,
                                             int ld, int lane) {
  const int m  = lane & 15;
  const int hh = lane >> 4;
  union { v16bf v; v8bf h[2]; } u;
  const bf16_t* p = base + (size_t)m * ld + hh * 8;
  u.h[0] = *(const v8bf*)(p);
  u.h[1] = *(const v8bf*)(p + 16);
  return u.v;
}

__device__ __forceinline__ v16bf load_frag_b(const bf16_t* __restrict__ base,
                                             int ld, int lane) {
  const int n  = lane & 15;
  const int hh = lane >> 4;
  union { v16bf v; v8bf h[2]; } u;
  const bf16_t* p = base + (size_t)n * ld + hh * 8;
  u.h[0] = *(const v8bf*)(p);
  u.h[1] = *(const v8bf*)(p + 16);
  return u.v;
}

__device__ __forceinline__ v8f wmma_bf16(v16bf a, v16bf b, v8f c) {
  // 8 args: (neg_a, A, neg_b, B, c_mod, C, reuse_a, reuse_b)
  return __builtin_amdgcn_wmma_f32_16x16x32_bf16(false, a, false, b,
                                                 (short)0, c, false, false);
}

// ---------------------------------------------------------------------------
// Kernel 1: weights f32 -> bf16 (concatenated [3][768][768])
// ---------------------------------------------------------------------------
__global__ __launch_bounds__(256) void wconv_kernel(
    const float* __restrict__ Wq, const float* __restrict__ Wk,
    const float* __restrict__ Wv, bf16_t* __restrict__ Wb) {
  const int total = 3 * D_MODEL * D_MODEL;
  int i = blockIdx.x * 256 + threadIdx.x;
  if (i < total) {
    const int z = i / (D_MODEL * D_MODEL);
    const int r = i - z * (D_MODEL * D_MODEL);
    const float* W = (z == 0) ? Wq : (z == 1) ? Wk : Wv;
    Wb[i] = (bf16_t)W[r];
  }
}

// ---------------------------------------------------------------------------
// Kernel 2: LayerNorm over D=768, output bf16. One block (256 thr) per row.
// ---------------------------------------------------------------------------
__global__ __launch_bounds__(256) void ln_kernel(
    const float* __restrict__ X, const float* __restrict__ gamma,
    const float* __restrict__ beta, bf16_t* __restrict__ Xh) {
  __shared__ float red[256];
  const int tid = threadIdx.x;
  const size_t base = (size_t)blockIdx.x * D_MODEL;

  const float v0 = X[base + tid];
  const float v1 = X[base + tid + 256];
  const float v2 = X[base + tid + 512];

  red[tid] = v0 + v1 + v2;
  __syncthreads();
  for (int off = 128; off > 0; off >>= 1) {
    if (tid < off) red[tid] += red[tid + off];
    __syncthreads();
  }
  const float mean = red[0] * (1.0f / D_MODEL);
  __syncthreads();

  const float d0 = v0 - mean, d1 = v1 - mean, d2 = v2 - mean;
  red[tid] = d0 * d0 + d1 * d1 + d2 * d2;
  __syncthreads();
  for (int off = 128; off > 0; off >>= 1) {
    if (tid < off) red[tid] += red[tid + off];
    __syncthreads();
  }
  const float inv = rsqrtf(red[0] * (1.0f / D_MODEL) + LN_EPS);

  Xh[base + tid      ] = (bf16_t)(d0 * inv * gamma[tid      ] + beta[tid      ]);
  Xh[base + tid + 256] = (bf16_t)(d1 * inv * gamma[tid + 256] + beta[tid + 256]);
  Xh[base + tid + 512] = (bf16_t)(d2 * inv * gamma[tid + 512] + beta[tid + 512]);
}

// ---------------------------------------------------------------------------
// Kernel 3: QKV projection  y = x @ W^T  (M=16384, N=768, K=768), bf16 WMMA.
// Block = 256 thr = 8 waves (2 M x 4 N), block tile 64x128, wave tile 32x32.
// z = 0:Q (row-major), 1:K (row-major), 2:V stored TRANSPOSED Vt[b][d][n].
// B operand B[k][n] = W[n][k] is contiguous in k in W's row-major storage, so
// fragments stream straight from global W (fully L2-resident).
// ---------------------------------------------------------------------------
__global__ __launch_bounds__(256) void qkv_gemm_kernel(
    const bf16_t* __restrict__ Xh, const bf16_t* __restrict__ Wall,
    bf16_t* __restrict__ Q, bf16_t* __restrict__ K, bf16_t* __restrict__ Vt) {
  const int z = blockIdx.z;
  const bf16_t* __restrict__ W = Wall + (size_t)z * D_MODEL * D_MODEL;

  const int lane = threadIdx.x & 31;
  const int wave = threadIdx.x >> 5;
  const int mBase = blockIdx.x * 64 + (wave >> 2) * 32;
  const int nBase = blockIdx.y * 128 + (wave & 3) * 32;

  v8f acc[2][2] = {{{0.f}, {0.f}}, {{0.f}, {0.f}}};

  for (int k = 0; k < D_MODEL; k += 32) {
    if (k + 32 < D_MODEL) {  // global_prefetch_b8 next K-slab of A rows
      __builtin_prefetch(Xh + (size_t)(mBase + (lane & 15)) * D_MODEL + k + 32, 0, 3);
    }
    v16bf a0 = load_frag_a(Xh + (size_t)(mBase     ) * D_MODEL + k, D_MODEL, lane);
    v16bf a1 = load_frag_a(Xh + (size_t)(mBase + 16) * D_MODEL + k, D_MODEL, lane);
    v16bf b0 = load_frag_b(W  + (size_t)(nBase     ) * D_MODEL + k, D_MODEL, lane);
    v16bf b1 = load_frag_b(W  + (size_t)(nBase + 16) * D_MODEL + k, D_MODEL, lane);
    acc[0][0] = wmma_bf16(a0, b0, acc[0][0]);
    acc[0][1] = wmma_bf16(a0, b1, acc[0][1]);
    acc[1][0] = wmma_bf16(a1, b0, acc[1][0]);
    acc[1][1] = wmma_bf16(a1, b1, acc[1][1]);
  }

  // C/D layout: VGPR e holds M = e + 8*(lane>>4), N = lane&15.
  #pragma unroll
  for (int mt = 0; mt < 2; ++mt) {
    #pragma unroll
    for (int nt = 0; nt < 2; ++nt) {
      #pragma unroll
      for (int e = 0; e < 8; ++e) {
        const int row = mBase + mt * 16 + e + ((lane >> 4) << 3);
        const int col = nBase + nt * 16 + (lane & 15);
        const bf16_t val = (bf16_t)acc[mt][nt][e];
        if (z == 0) {
          Q[(size_t)row * D_MODEL + col] = val;
        } else if (z == 1) {
          K[(size_t)row * D_MODEL + col] = val;
        } else {  // V transposed per batch: Vt[b][col][n]
          const int b  = row >> 11;        // /2048
          const int nn = row & (SEQ - 1);  // %2048
          Vt[((size_t)b * D_MODEL + col) * SEQ + nn] = val;
        }
      }
    }
  }
}

// ---------------------------------------------------------------------------
// Kernel 4: flash attention. Block = (batch, 16-query tile), 256 thr / 8 waves.
// Key chunks of 256 (wave w scores keys w*32..w*32+31); online softmax in LDS;
// each wave accumulates a 96-column slice of ctx (6 x v8f), rescaled per chunk.
// ---------------------------------------------------------------------------
#define CHUNK 256
#define QS_LD 776   // 768 + 8 pad: 16B/row bank shift -> conflict-free frags
#define SB_LD 260
#define PB_LD 264   // 256 + 8 pad, keeps 16-B alignment for v8bf loads

__global__ __launch_bounds__(256) void attn_kernel(
    const bf16_t* __restrict__ Q, const bf16_t* __restrict__ Km,
    const bf16_t* __restrict__ Vt, const unsigned char* __restrict__ mask,
    float* __restrict__ out) {
  __shared__ bf16_t Qs[16 * QS_LD];
  __shared__ float  Sb[16 * SB_LD];
  __shared__ bf16_t Pb[16 * PB_LD];
  __shared__ float  red[16][16];
  __shared__ float  rowM[16], rowL[16], rowC[16];
  __shared__ unsigned char mk[CHUNK];

  const int b   = blockIdx.y;
  const int q0  = blockIdx.x * 16;
  const int tid = threadIdx.x;
  const int lane = tid & 31;
  const int wave = tid >> 5;
  const size_t qBase = ((size_t)b * SEQ + q0) * D_MODEL;
  const size_t kRow0 = (size_t)b * SEQ;

  // Stage Q tile (16 x 768 bf16) into LDS: 16 threads/row x 48 elems (6x v8bf)
  {
    const int r = tid >> 4;
    const int c = (tid & 15) * 48;
    const bf16_t* src = Q + qBase + (size_t)r * D_MODEL + c;
    bf16_t* dst = &Qs[r * QS_LD + c];
    #pragma unroll
    for (int j = 0; j < 6; ++j)
      *(v8bf*)(dst + j * 8) = *(const v8bf*)(src + j * 8);
  }
  if (tid < 16) { rowM[tid] = -1e30f; rowL[tid] = 0.f; }

  v8f acc[6] = {{0.f}, {0.f}, {0.f}, {0.f}, {0.f}, {0.f}};
  const int c0 = wave * 32;  // this wave's key offset within a chunk

  for (int kBase = 0; kBase < SEQ; kBase += CHUNK) {
    __syncthreads();                       // prev-iter Pb/mk consumers done
    mk[tid] = mask[kRow0 + kBase + tid];   // 256 mask bytes for this chunk
    __syncthreads();                       // Qs (first iter) + mk visible

    // ---- S = Q K^T for this wave's 32 keys (2 N-tiles, K-loop 768/32) ----
    v8f s[2] = {{0.f}, {0.f}};
    for (int k = 0; k < D_MODEL; k += 32) {
      v16bf a  = load_frag_a(&Qs[k], QS_LD, lane);
      v16bf b0 = load_frag_b(Km + (kRow0 + kBase + c0     ) * D_MODEL + k, D_MODEL, lane);
      v16bf b1 = load_frag_b(Km + (kRow0 + kBase + c0 + 16) * D_MODEL + k, D_MODEL, lane);
      s[0] = wmma_bf16(a, b0, s[0]);
      s[1] = wmma_bf16(a, b1, s[1]);
    }
    #pragma unroll
    for (int nt = 0; nt < 2; ++nt) {
      #pragma unroll
      for (int e = 0; e < 8; ++e) {
        const int r = e + ((lane >> 4) << 3);
        const int c = c0 + nt * 16 + (lane & 15);
        float v = s[nt][e] * ATT_SCALE;
        if (mk[c]) v = -1e30f;             // masked key -> -inf (softmax -> 0)
        Sb[r * SB_LD + c] = v;
      }
    }
    __syncthreads();

    // ---- online softmax over the 16x256 chunk: 16 threads per row ----
    const int r   = tid >> 4;
    const int s16 = tid & 15;
    float mloc = -1e30f;
    #pragma unroll
    for (int j = 0; j < 16; ++j)
      mloc = fmaxf(mloc, Sb[r * SB_LD + s16 * 16 + j]);
    red[r][s16] = mloc;
    __syncthreads();
    if (s16 == 0) {
      float cm = red[r][0];
      #pragma unroll
      for (int j = 1; j < 16; ++j) cm = fmaxf(cm, red[r][j]);
      const float mOld = rowM[r];
      const float mNew = fmaxf(mOld, cm);
      rowC[r] = __expf(mOld - mNew);
      rowM[r] = mNew;
    }
    __syncthreads();
    const float mNew = rowM[r];
    float sloc = 0.f;
    #pragma unroll
    for (int j = 0; j < 16; ++j) {
      const int c = s16 * 16 + j;
      const float p = __expf(Sb[r * SB_LD + c] - mNew);
      sloc += p;
      Pb[r * PB_LD + c] = (bf16_t)p;
    }
    red[r][s16] = sloc;
    __syncthreads();
    if (s16 == 0) {
      float t = 0.f;
      #pragma unroll
      for (int j = 0; j < 16; ++j) t += red[r][j];
      rowL[r] = rowL[r] * rowC[r] + t;
    }
    __syncthreads();

    // ---- rescale running ctx accumulators by per-row correction ----
    float cv[8];
    #pragma unroll
    for (int e = 0; e < 8; ++e) cv[e] = rowC[e + ((lane >> 4) << 3)];
    #pragma unroll
    for (int nt = 0; nt < 6; ++nt)
      #pragma unroll
      for (int e = 0; e < 8; ++e) acc[nt][e] *= cv[e];

    // ---- ctx += P @ V : this wave owns d-columns [wave*96, wave*96+96) ----
    for (int ks = 0; ks < CHUNK; ks += 32) {
      v16bf a = load_frag_a(&Pb[ks], PB_LD, lane);
      #pragma unroll
      for (int nt = 0; nt < 6; ++nt) {
        const int dcol = wave * 96 + nt * 16;
        const bf16_t* vb = Vt + ((size_t)b * D_MODEL + dcol) * SEQ + kBase + ks;
        v16bf bb = load_frag_b(vb, SEQ, lane);
        acc[nt] = wmma_bf16(a, bb, acc[nt]);
      }
    }
  }
  __syncthreads();

  // ---- normalize by row sums and write ctx (f32) ----
  float inv[8];
  #pragma unroll
  for (int e = 0; e < 8; ++e) {
    const float l = rowL[e + ((lane >> 4) << 3)];
    inv[e] = 1.0f / ((l > 0.f) ? l : 1.0f);
  }
  #pragma unroll
  for (int nt = 0; nt < 6; ++nt) {
    #pragma unroll
    for (int e = 0; e < 8; ++e) {
      const int row = q0 + e + ((lane >> 4) << 3);
      const int col = wave * 96 + nt * 16 + (lane & 15);
      out[((size_t)b * SEQ + row) * D_MODEL + col] = acc[nt][e] * inv[e];
    }
  }
}

// ---------------------------------------------------------------------------
// Host launcher.  Inputs: 0 features f32[8,2048,768], 1 mask bool[8,2048],
// 2..4 Wq/Wk/Wv f32[768,768], 5 gamma, 6 beta.  Output: ctx f32[8,2048,768].
// Workspace: xhat | Wbf16[3] | Q | K | Vt   (~100 MB, all bf16)
// ---------------------------------------------------------------------------
extern "C" void kernel_launch(void* const* d_in, const int* in_sizes, int n_in,
                              void* d_out, int out_size, void* d_ws, size_t ws_size,
                              hipStream_t stream) {
  const float* features = (const float*)d_in[0];
  const unsigned char* mask = (const unsigned char*)d_in[1];
  const float* Wq = (const float*)d_in[2];
  const float* Wk = (const float*)d_in[3];
  const float* Wv = (const float*)d_in[4];
  const float* gamma = (const float*)d_in[5];
  const float* beta  = (const float*)d_in[6];
  float* out = (float*)d_out;

  bf16_t* xh = (bf16_t*)d_ws;
  bf16_t* wb = xh + (size_t)ROWS * D_MODEL;
  bf16_t* q  = wb + (size_t)3 * D_MODEL * D_MODEL;
  bf16_t* k  = q  + (size_t)ROWS * D_MODEL;
  bf16_t* vt = k  + (size_t)ROWS * D_MODEL;

  const int wtotal = 3 * D_MODEL * D_MODEL;
  wconv_kernel<<<(wtotal + 255) / 256, 256, 0, stream>>>(Wq, Wk, Wv, wb);
  ln_kernel<<<ROWS, 256, 0, stream>>>(features, gamma, beta, xh);
  qkv_gemm_kernel<<<dim3(ROWS / 64, D_MODEL / 128, 3), 256, 0, stream>>>(xh, wb, q, k, vt);
  attn_kernel<<<dim3(SEQ / 16, BATCH), 256, 0, stream>>>(q, k, vt, mask, out);
}
